// OscillatorAttention_18408229830882
// MI455X (gfx1250) — matrix operand order repaired and checked by reading, not compiled, not run on hardware
//
#include <hip/hip_runtime.h>

// ---------------- problem constants ----------------
constexpr int B_  = 4;
constexpr int T_  = 1024;
constexpr int DM_ = 1024;
constexpr int H_  = 16;
constexpr int DK_ = 64;
constexpr int NSTEPS_ = 3;

// ---------------- WMMA types ----------------
typedef __attribute__((ext_vector_type(16))) __bf16 v16bf;
typedef __attribute__((ext_vector_type(8)))  __bf16 v8bf;
typedef __attribute__((ext_vector_type(8)))  float  v8f;

// ---------------- bf16 helpers (bit-level, no header dependency) ----------------
__device__ __forceinline__ unsigned short f32_to_bf16(float f) {
    unsigned int u = __float_as_uint(f);
    u += 0x7FFFu + ((u >> 16) & 1u);   // round-to-nearest-even
    return (unsigned short)(u >> 16);
}
__device__ __forceinline__ float bf16_to_f32(unsigned short h) {
    return __uint_as_float(((unsigned int)h) << 16);
}

__device__ __forceinline__ void store_val(float* p, float v)          { *p = v; }
__device__ __forceinline__ void store_val(unsigned short* p, float v) { *p = f32_to_bf16(v); }

// ---------------- fragment loads (wave32, contiguous k = 16*(lane/16) + s, same order for A and B) ----------------
// A: 16x32 bf16. lanes 0-15: row=lane, K = 0..15; lanes 16-31: row=lane-16, K = 16..31
__device__ __forceinline__ v16bf load_frag_a(const unsigned short* base, int ld, int lane) {
    int row = lane & 15;
    int kh  = (lane >> 4) << 4;                   // 0 or 16
    const unsigned short* p = base + (long long)row * ld + kh;
    v8bf lo = *reinterpret_cast<const v8bf*>(p);          // K = kh .. kh+7
    v8bf hi = *reinterpret_cast<const v8bf*>(p + 8);      // K = kh+8 .. kh+15
    v16bf r;
#pragma unroll
    for (int i = 0; i < 8; ++i) { r[i] = lo[i]; r[i + 8] = hi[i]; }
    return r;
}
// B: 32x16 bf16, supplied as B^T (N x K row-major). lanes 0-15: col=lane, K=0..15; lanes 16-31: col=lane-16, K=16..31
__device__ __forceinline__ v16bf load_frag_b(const unsigned short* bt, int ldbt, int lane) {
    int n  = lane & 15;
    int k0 = (lane >> 4) << 4;                    // 0 or 16
    const unsigned short* p = bt + (long long)n * ldbt + k0;
    v8bf lo = *reinterpret_cast<const v8bf*>(p);
    v8bf hi = *reinterpret_cast<const v8bf*>(p + 8);
    v16bf r;
#pragma unroll
    for (int i = 0; i < 8; ++i) { r[i] = lo[i]; r[i + 8] = hi[i]; }
    return r;
}

// ---------------- elementwise converts ----------------
__global__ __launch_bounds__(256) void cvt_f32_bf16_kernel(const float* __restrict__ src,
                                                           unsigned short* __restrict__ dst,
                                                           long long n) {
    long long i = (long long)blockIdx.x * blockDim.x + threadIdx.x;
    if (i < n) dst[i] = f32_to_bf16(src[i]);
}

// src: batch of (rows x cols) f32 row-major -> dst: batch of (cols x rows) bf16. blockIdx.y = batch.
__global__ __launch_bounds__(256) void cvt_transpose_kernel(const float* __restrict__ src,
                                                            unsigned short* __restrict__ dst,
                                                            int rows, int cols) {
    long long n    = (long long)rows * cols;
    long long base = (long long)blockIdx.y * n;
    long long i    = (long long)blockIdx.x * blockDim.x + threadIdx.x;
    if (i >= n) return;
    int r = (int)(i / cols), c = (int)(i % cols);
    dst[base + (long long)c * rows + r] = f32_to_bf16(src[base + i]);
}

// ---------------- generic batched WMMA GEMM ----------------
// C[b] (M x N, ldc) = alpha * A[b](M x K, lda, bf16) * Bt[b](N x K, ldbt, bf16)^T  [+ beta * Cadd[b]]
// if lam_ptr: alpha = sigmoid(*lam_ptr), beta = 1 - alpha  (fused topology-propagation step)
// if head_layout: C index (m = b*T + t, n = h*DK + dk) remapped to compact (B,H,T,DK); Cadd ignored.
template <typename OutT>
__global__ __launch_bounds__(256) void gemm_bf16_kernel(
        const unsigned short* __restrict__ A,  long long sA,  int lda,
        const unsigned short* __restrict__ Bt, long long sB,  int ldbt,
        OutT* __restrict__ C,                  long long sC,  int ldc,
        const float* __restrict__ Cadd,        long long sCadd,
        int M, int N, int K, float alpha, float beta,
        const float* __restrict__ lam_ptr, int head_layout) {
    int batch = blockIdx.y;
    const unsigned short* Ab = A  + (long long)batch * sA;
    const unsigned short* Bb = Bt + (long long)batch * sB;
    OutT* Cb = C + (long long)batch * sC;
    int lane   = threadIdx.x & 31;
    int wid    = blockIdx.x * (blockDim.x >> 5) + (threadIdx.x >> 5);
    int tilesN = N >> 4;
    int tilesM = M >> 4;
    if (wid >= tilesM * tilesN) return;                 // wave-uniform: EXEC all-ones at WMMA
    int tm = wid / tilesN, tn = wid % tilesN;
    if (lam_ptr) {
        float l = 1.0f / (1.0f + __expf(-lam_ptr[0]));
        alpha = l; beta = 1.0f - l;
    }
    const unsigned short* Abase = Ab + (long long)(tm * 16) * lda;
    const unsigned short* Bbase = Bb + (long long)(tn * 16) * ldbt;
    v8f acc = {};
    for (int k0 = 0; k0 < K; k0 += 32) {
        v16bf a = load_frag_a(Abase + k0, lda, lane);
        v16bf b = load_frag_b(Bbase + k0, ldbt, lane);
        acc = __builtin_amdgcn_wmma_f32_16x16x32_bf16(false, a, false, b, (short)0, acc, false, false);
    }
    int col   = tn * 16 + (lane & 15);
    int rbase = tm * 16 + ((lane >> 4) << 3);
    const float* CaddB = Cadd ? (Cadd + (long long)batch * sCadd) : nullptr;
#pragma unroll
    for (int r = 0; r < 8; ++r) {
        int row = rbase + r;
        float v = alpha * acc[r];
        if (head_layout) {
            int bb = row / T_, t = row % T_;
            int hh = col / DK_, dk = col % DK_;
            long long idx = (((long long)(bb * H_ + hh)) * T_ + t) * DK_ + dk;
            store_val(&Cb[idx], v);
        } else {
            long long idx = (long long)row * ldc + col;
            if (CaddB) v += beta * CaddB[idx];
            store_val(&Cb[idx], v);
        }
    }
}

// ---------------- fused scores (Q K^T / sqrt(DK)) + row softmax -> bf16 A ----------------
// grid: (T/16, B*H), block 256 (8 waves). Each block: 16 rows x full 1024 cols.
__global__ __launch_bounds__(256) void scores_softmax_kernel(
        const unsigned short* __restrict__ Qb,   // (B,T,DM) bf16
        const unsigned short* __restrict__ Kb,   // (B,T,DM) bf16
        unsigned short* __restrict__ Aout) {     // (B,H,T,T) bf16
    __shared__ float sc[16][T_ + 4];
    int bh = blockIdx.y;
    int b = bh / H_, h = bh % H_;
    int t0   = blockIdx.x * 16;
    int lane = threadIdx.x & 31;
    int wave = threadIdx.x >> 5;                 // 0..7
    const unsigned short* Qs = Qb + (long long)b * T_ * DM_ + h * DK_;
    const unsigned short* Ks = Kb + (long long)b * T_ * DM_ + h * DK_;
    const unsigned short* Arow0 = Qs + (long long)t0 * DM_;
    // 8 waves x 8 column-tiles each = 64 tiles = 1024 cols
    for (int it = 0; it < 8; ++it) {
        int n0 = (wave * 8 + it) * 16;
        v8f acc = {};
#pragma unroll
        for (int k0 = 0; k0 < DK_; k0 += 32) {
            v16bf a = load_frag_a(Arow0 + k0, DM_, lane);
            v16bf bb = load_frag_b(Ks + (long long)n0 * DM_ + k0, DM_, lane);
            acc = __builtin_amdgcn_wmma_f32_16x16x32_bf16(false, a, false, bb, (short)0, acc, false, false);
        }
        int col   = n0 + (lane & 15);
        int rbase = (lane >> 4) << 3;
#pragma unroll
        for (int r = 0; r < 8; ++r)
            sc[rbase + r][col] = acc[r] * 0.125f;     // 1/sqrt(64)
    }
    __syncthreads();
    // softmax: each wave owns rows 2*wave, 2*wave+1
    for (int rr = 0; rr < 2; ++rr) {
        int row = wave * 2 + rr;
        float mx = -__builtin_inff();
        for (int c = lane; c < T_; c += 32) mx = fmaxf(mx, sc[row][c]);
        for (int off = 16; off > 0; off >>= 1) mx = fmaxf(mx, __shfl_xor(mx, off, 32));
        float sum = 0.0f;
        for (int c = lane; c < T_; c += 32) sum += __expf(sc[row][c] - mx);
        for (int off = 16; off > 0; off >>= 1) sum += __shfl_xor(sum, off, 32);
        float inv = 1.0f / sum;
        unsigned short* Ar = Aout + (((long long)bh * T_) + (t0 + row)) * T_;
        for (int c = lane; c < T_; c += 32)
            Ar[c] = f32_to_bf16(__expf(sc[row][c] - mx) * inv);
    }
}

// ---------------- mean over T of complex D, per (b,h,dk) ----------------
__global__ __launch_bounds__(256) void mean_kernel(const float* __restrict__ Dre,
                                                   const float* __restrict__ Dim,
                                                   float* __restrict__ Mre,
                                                   float* __restrict__ Mim) {
    int bh  = blockIdx.x;
    int tid = threadIdx.x;           // 256
    int dk  = tid & 63;
    int sl  = tid >> 6;              // 0..3
    const float* re = Dre + (long long)bh * T_ * DK_;
    const float* im = Dim + (long long)bh * T_ * DK_;
    float sre = 0.f, sim = 0.f;
    for (int t = sl * (T_ / 4); t < (sl + 1) * (T_ / 4); ++t) {
        sre += re[(long long)t * DK_ + dk];
        sim += im[(long long)t * DK_ + dk];
    }
    __shared__ float shre[256], shim[256];
    shre[tid] = sre; shim[tid] = sim;
    __syncthreads();
    if (tid < 64) {
        float tre = shre[tid] + shre[tid + 64] + shre[tid + 128] + shre[tid + 192];
        float tim = shim[tid] + shim[tid + 64] + shim[tid + 128] + shim[tid + 192];
        Mre[bh * DK_ + tid] = tre * (1.0f / T_);
        Mim[bh * DK_ + tid] = tim * (1.0f / T_);
    }
}

// ---------------- gate softmax over T + V gating ----------------
// one block per (bh*DK + dk) column; 128 threads, 8 t's each
__global__ __launch_bounds__(128) void gate_kernel(
        const float* __restrict__ Dre, const float* __restrict__ Dim,
        const float* __restrict__ Mre, const float* __restrict__ Mim,
        const float* __restrict__ Vproj,            // (B,T,DM) f32
        float* __restrict__ gate_out,               // (B,H,T,DK) f32
        unsigned short* __restrict__ gatedV) {      // (B,T,DM) bf16
    int col = blockIdx.x;                 // bh*DK + dk
    int bh = col / DK_, dk = col % DK_;
    int b = bh / H_, h = bh % H_;
    float mre = Mre[col], mim = Mim[col];
    float mabs = sqrtf(mre * mre + mim * mim);
    const float* re = Dre + (long long)bh * T_ * DK_ + dk;
    const float* im = Dim + (long long)bh * T_ * DK_ + dk;
    int tid = threadIdx.x;                // 128
    float g[8];
    float mx = -__builtin_inff();
#pragma unroll
    for (int i = 0; i < 8; ++i) {
        int t = i * 128 + tid;
        float dre = re[(long long)t * DK_], dim = im[(long long)t * DK_];
        float dot = dre * mre + dim * mim;
        float nrm = sqrtf(dre * dre + dim * dim) * mabs + 1e-8f;
        g[i] = dot / nrm;
        mx = fmaxf(mx, g[i]);
    }
    __shared__ float sh[128];
    sh[tid] = mx; __syncthreads();
    for (int o = 64; o > 0; o >>= 1) { if (tid < o) sh[tid] = fmaxf(sh[tid], sh[tid + o]); __syncthreads(); }
    mx = sh[0]; __syncthreads();
    float s = 0.f;
#pragma unroll
    for (int i = 0; i < 8; ++i) { g[i] = __expf(g[i] - mx); s += g[i]; }
    sh[tid] = s; __syncthreads();
    for (int o = 64; o > 0; o >>= 1) { if (tid < o) sh[tid] += sh[tid + o]; __syncthreads(); }
    float inv = 1.0f / sh[0];
#pragma unroll
    for (int i = 0; i < 8; ++i) {
        int t = i * 128 + tid;
        float gg = g[i] * inv;
        long long gi = ((long long)bh * T_ + t) * DK_ + dk;
        gate_out[gi] = gg;
        long long vi = ((long long)b * T_ + t) * DM_ + h * DK_ + dk;
        gatedV[vi] = f32_to_bf16(gg * Vproj[vi]);
    }
}

// ---------------- launcher ----------------
static inline size_t align256(size_t x) { return (x + 255) & ~(size_t)255; }

extern "C" void kernel_launch(void* const* d_in, const int* in_sizes, int n_in,
                              void* d_out, int out_size, void* d_ws, size_t ws_size,
                              hipStream_t stream) {
    (void)in_sizes; (void)n_in; (void)out_size; (void)ws_size;
    const float* x     = (const float*)d_in[0];
    const float* Ws[6] = { (const float*)d_in[1], (const float*)d_in[2], (const float*)d_in[3],
                           (const float*)d_in[4], (const float*)d_in[5], (const float*)d_in[6] };
    const float* logit_lam = (const float*)d_in[7];

    const size_t nBTDM = (size_t)B_ * T_ * DM_;          // 4 Mi elems
    const size_t nW    = (size_t)DM_ * DM_;              // 1 Mi
    const size_t nBHTT = (size_t)B_ * H_ * T_ * T_;      // 64 Mi

    float* out  = (float*)d_out;                         // (B,T,DM)
    float* gate = (float*)d_out + nBTDM;                 // (B,H,T,DK)

    char* ws = (char*)d_ws;
    size_t off = 0;
    auto alloc = [&](size_t bytes) -> char* { char* p = ws + off; off = align256(off + bytes); return p; };

    unsigned short* xb   = (unsigned short*)alloc(nBTDM * 2);
    unsigned short* Wt[6];
    for (int i = 0; i < 6; ++i) Wt[i] = (unsigned short*)alloc(nW * 2);
    unsigned short* Qb   = (unsigned short*)alloc(nBTDM * 2);
    unsigned short* Kb   = (unsigned short*)alloc(nBTDM * 2);
    float*          Vf   = (float*)alloc(nBTDM * 4);
    float*          D0re = (float*)alloc(nBTDM * 4);
    float*          D0im = (float*)alloc(nBTDM * 4);
    float*          D1re = (float*)alloc(nBTDM * 4);
    float*          D1im = (float*)alloc(nBTDM * 4);
    unsigned short* Dtre = (unsigned short*)alloc(nBTDM * 2);
    unsigned short* Dtim = (unsigned short*)alloc(nBTDM * 2);
    unsigned short* Abuf = (unsigned short*)alloc(nBHTT * 2);
    float*          Mre  = (float*)alloc((size_t)B_ * H_ * DK_ * 4);
    float*          Mim  = (float*)alloc((size_t)B_ * H_ * DK_ * 4);
    unsigned short* gVb  = (unsigned short*)alloc(nBTDM * 2);

    // 1) x -> bf16 ; weights -> bf16 transposed (B^T layout: N x K)
    cvt_f32_bf16_kernel<<<dim3((unsigned)(nBTDM / 256)), 256, 0, stream>>>(x, xb, (long long)nBTDM);
    for (int i = 0; i < 6; ++i)
        cvt_transpose_kernel<<<dim3((unsigned)(nW / 256), 1), 256, 0, stream>>>(Ws[i], Wt[i], DM_, DM_);

    // 2) projections: waves = (4096/16)*(1024/16) = 16384 -> 2048 blocks
    const unsigned projBlocks = ((B_ * T_ / 16) * (DM_ / 16)) / 8;
    gemm_bf16_kernel<unsigned short><<<dim3(projBlocks, 1), 256, 0, stream>>>(
        xb, 0, DM_, Wt[0], 0, DM_, Qb, 0, DM_, nullptr, 0,
        B_ * T_, DM_, DM_, 1.0f, 0.0f, nullptr, 0);
    gemm_bf16_kernel<unsigned short><<<dim3(projBlocks, 1), 256, 0, stream>>>(
        xb, 0, DM_, Wt[1], 0, DM_, Kb, 0, DM_, nullptr, 0,
        B_ * T_, DM_, DM_, 1.0f, 0.0f, nullptr, 0);
    gemm_bf16_kernel<float><<<dim3(projBlocks, 1), 256, 0, stream>>>(
        xb, 0, DM_, Wt[2], 0, DM_, D0re, 0, DM_, nullptr, 0,
        B_ * T_, DM_, DM_, 1.0f, 0.0f, nullptr, 1);          // head-compact (B,H,T,DK)
    gemm_bf16_kernel<float><<<dim3(projBlocks, 1), 256, 0, stream>>>(
        xb, 0, DM_, Wt[3], 0, DM_, D0im, 0, DM_, nullptr, 0,
        B_ * T_, DM_, DM_, 1.0f, 0.0f, nullptr, 1);
    gemm_bf16_kernel<float><<<dim3(projBlocks, 1), 256, 0, stream>>>(
        xb, 0, DM_, Wt[4], 0, DM_, Vf, 0, DM_, nullptr, 0,
        B_ * T_, DM_, DM_, 1.0f, 0.0f, nullptr, 0);

    // 3) fused scores + softmax -> A (bf16, 128 MB, L2-resident)
    scores_softmax_kernel<<<dim3(T_ / 16, B_ * H_), 256, 0, stream>>>(Qb, Kb, Abuf);

    // 4) topology propagation: D <- (1-lam) D + lam A D, 3 steps, complex via re/im planes
    float* curRe = D0re; float* curIm = D0im; float* nxtRe = D1re; float* nxtIm = D1im;
    const unsigned propBlocks = ((T_ / 16) * (DK_ / 16)) / 8;  // 256 waves -> 32 blocks, grid.y = 64
    for (int s = 0; s < NSTEPS_; ++s) {
        cvt_transpose_kernel<<<dim3((T_ * DK_) / 256, B_ * H_), 256, 0, stream>>>(curRe, Dtre, T_, DK_);
        cvt_transpose_kernel<<<dim3((T_ * DK_) / 256, B_ * H_), 256, 0, stream>>>(curIm, Dtim, T_, DK_);
        gemm_bf16_kernel<float><<<dim3(propBlocks, B_ * H_), 256, 0, stream>>>(
            Abuf, (long long)T_ * T_, T_, Dtre, (long long)DK_ * T_, T_,
            nxtRe, (long long)T_ * DK_, DK_, curRe, (long long)T_ * DK_,
            T_, DK_, T_, 0.5f, 0.5f, logit_lam, 0);
        gemm_bf16_kernel<float><<<dim3(propBlocks, B_ * H_), 256, 0, stream>>>(
            Abuf, (long long)T_ * T_, T_, Dtim, (long long)DK_ * T_, T_,
            nxtIm, (long long)T_ * DK_, DK_, curIm, (long long)T_ * DK_,
            T_, DK_, T_, 0.5f, 0.5f, logit_lam, 0);
        float* tr = curRe; curRe = nxtRe; nxtRe = tr;
        float* ti = curIm; curIm = nxtIm; nxtIm = ti;
    }

    // 5) readout: mean field, gate softmax over T, gated V (bf16)
    mean_kernel<<<dim3(B_ * H_), 256, 0, stream>>>(curRe, curIm, Mre, Mim);
    gate_kernel<<<dim3(B_ * H_ * DK_), 128, 0, stream>>>(curRe, curIm, Mre, Mim, Vf, gate, gVb);

    // 6) output projection: out = gatedV @ W_O
    gemm_bf16_kernel<float><<<dim3(projBlocks, 1), 256, 0, stream>>>(
        gVb, 0, DM_, Wt[5], 0, DM_, out, 0, DM_, nullptr, 0,
        B_ * T_, DM_, DM_, 1.0f, 0.0f, nullptr, 0);
}
